// PointNetSetAbstraction_24120536335102
// MI455X (gfx1250) — matrix operand order repaired and checked by reading, not compiled, not run on hardware
//
#include <hip/hip_runtime.h>
#include <math.h>

// Problem constants (match reference)
#define BB      16
#define NN      8192
#define NP      512
#define NS      32
#define RAD2    0.16f        // 0.4^2
#define EPS_BN  1e-5f

typedef __attribute__((ext_vector_type(16))) _Float16 v16h;
typedef __attribute__((ext_vector_type(8)))  _Float16 v8h;
typedef __attribute__((ext_vector_type(8)))  float    v8f;

__device__ __forceinline__ v8f wmma_f16(v16h a, v16h b, v8f c) {
  // D = A(16x32 f16) x B(32x16 f16) + C(16x16 f32)
  return __builtin_amdgcn_wmma_f32_16x16x32_f16(
      /*neg_a=*/false, a, /*neg_b=*/false, b,
      /*c_mod=*/(short)0, c, /*reuse_a=*/false, /*reuse_b=*/false);
}

// ---------------------------------------------------------------------------
// Prep kernel (1 block): pre-swizzle weights into WMMA B-fragment order (f16,
// zero-padded) and fold conv-bias + BN into per-channel (scale, shift).
// B fragment layout (32x16 f16): lane l -> col (l&15); half h -> K=(h>>3)*16+(l>>4)*8+(h&7)
// Flat index for fragment (j, kt): ((j*KT + kt)*32 + lane)*16 + h
// ---------------------------------------------------------------------------
__global__ __launch_bounds__(256) void prep_kernel(
    const float* __restrict__ w0, const float* __restrict__ cb0,
    const float* __restrict__ g0, const float* __restrict__ be0,
    const float* __restrict__ m0, const float* __restrict__ v0,
    const float* __restrict__ w1, const float* __restrict__ cb1,
    const float* __restrict__ g1, const float* __restrict__ be1,
    const float* __restrict__ m1, const float* __restrict__ v1,
    const float* __restrict__ w2, const float* __restrict__ cb2,
    const float* __restrict__ g2, const float* __restrict__ be2,
    const float* __restrict__ m2, const float* __restrict__ v2,
    _Float16* __restrict__ f0, _Float16* __restrict__ f1,
    _Float16* __restrict__ f2,
    float2* __restrict__ bn0, float2* __restrict__ bn1,
    float2* __restrict__ bn2)
{
  const int tid = threadIdx.x;

  // layer 0: 4 n-tiles, 1 k-tile (Cin=9 padded to 32) -> 2048 halves
  for (int i = tid; i < 4 * 32 * 16; i += 256) {
    const int h = i & 15, lane = (i >> 4) & 31, j = i >> 9;
    const int oc = j * 16 + (lane & 15);
    const int k  = ((h >> 3) << 4) + ((lane >> 4) << 3) + (h & 7);
    f0[i] = (k < 9) ? (_Float16)w0[oc * 9 + k] : (_Float16)0.f;
  }
  // layer 1: 4 n-tiles, 2 k-tiles (Cin=64) -> 4096 halves
  for (int i = tid; i < 8 * 32 * 16; i += 256) {
    const int h = i & 15, lane = (i >> 4) & 31;
    const int kt = (i >> 9) & 1, j = i >> 10;
    const int oc = j * 16 + (lane & 15);
    const int k  = kt * 32 + ((h >> 3) << 4) + ((lane >> 4) << 3) + (h & 7);
    f1[i] = (_Float16)w1[oc * 64 + k];
  }
  // layer 2: 8 n-tiles, 2 k-tiles (Cin=64) -> 8192 halves
  for (int i = tid; i < 16 * 32 * 16; i += 256) {
    const int h = i & 15, lane = (i >> 4) & 31;
    const int kt = (i >> 9) & 1, j = i >> 10;
    const int oc = j * 16 + (lane & 15);
    const int k  = kt * 32 + ((h >> 3) << 4) + ((lane >> 4) << 3) + (h & 7);
    f2[i] = (_Float16)w2[oc * 64 + k];
  }
  // fused BN: y = x*sc + sh, sc = g/sqrt(v+eps), sh = cb*sc + be - m*sc
  for (int i = tid; i < 64; i += 256) {
    const float sc = g0[i] / sqrtf(v0[i] + EPS_BN);
    bn0[i] = make_float2(sc, cb0[i] * sc + be0[i] - m0[i] * sc);
  }
  for (int i = tid; i < 64; i += 256) {
    const float sc = g1[i] / sqrtf(v1[i] + EPS_BN);
    bn1[i] = make_float2(sc, cb1[i] * sc + be1[i] - m1[i] * sc);
  }
  for (int i = tid; i < 128; i += 256) {
    const float sc = g2[i] / sqrtf(v2[i] + EPS_BN);
    bn2[i] = make_float2(sc, cb2[i] * sc + be2[i] - m2[i] * sc);
  }
}

// ---------------------------------------------------------------------------
// Kernel 1: Farthest point sampling. One block per batch, 256 threads.
// Each thread owns 32 points in registers. 512 sequential argmax rounds.
// ---------------------------------------------------------------------------
__global__ __launch_bounds__(256) void fps_kernel(
    const float* __restrict__ xyz,   // (B,3,N)
    float* __restrict__ newXyz,      // ws: (B,NP,3)
    float* __restrict__ out1)        // (B,3,NP)
{
  const int b   = blockIdx.x;
  const int tid = threadIdx.x;
  const float* xb = xyz + (size_t)b * 3 * NN;

  float px[32], py[32], pz[32], pd[32];
#pragma unroll
  for (int j = 0; j < 32; ++j) {
    const int p = j * 256 + tid;
    px[j] = xb[p];
    py[j] = xb[NN + p];
    pz[j] = xb[2 * NN + p];
    pd[j] = 1e10f;
  }

  __shared__ float sVal[256];
  __shared__ int   sIdx[256];
  __shared__ int   sFar;
  if (tid == 0) sFar = 0;
  __syncthreads();

  for (int it = 0; it < NP; ++it) {
    const int far = sFar;
    const float cx = xb[far];
    const float cy = xb[NN + far];
    const float cz = xb[2 * NN + far];
    if (tid == 0) {
      newXyz[((size_t)b * NP + it) * 3 + 0] = cx;
      newXyz[((size_t)b * NP + it) * 3 + 1] = cy;
      newXyz[((size_t)b * NP + it) * 3 + 2] = cz;
      out1[((size_t)b * 3 + 0) * NP + it] = cx;
      out1[((size_t)b * 3 + 1) * NP + it] = cy;
      out1[((size_t)b * 3 + 2) * NP + it] = cz;
    }
    float best = -1.0f;
    int   bi   = 0x7fffffff;
#pragma unroll
    for (int j = 0; j < 32; ++j) {
      const float dx = px[j] - cx;
      const float dy = py[j] - cy;
      const float dz = pz[j] - cz;
      const float d  = dx * dx + dy * dy + dz * dz;
      pd[j] = fminf(pd[j], d);
      if (pd[j] > best) { best = pd[j]; bi = j * 256 + tid; }
    }
    sVal[tid] = best;
    sIdx[tid] = bi;
    __syncthreads();
    for (int st = 128; st > 0; st >>= 1) {
      if (tid < st) {
        const float v2 = sVal[tid + st];
        const int   i2 = sIdx[tid + st];
        if (v2 > sVal[tid] || (v2 == sVal[tid] && i2 < sIdx[tid])) {
          sVal[tid] = v2; sIdx[tid] = i2;
        }
      }
      __syncthreads();
    }
    if (tid == 0) sFar = sIdx[0];
    __syncthreads();
  }
}

// ---------------------------------------------------------------------------
// Kernel 2: Ball query. One wave32 per (b,s). Ballot + prefix-popcount
// compaction collects the 32 smallest in-radius indices in ascending order.
// ---------------------------------------------------------------------------
__global__ __launch_bounds__(128) void ballq_kernel(
    const float* __restrict__ xyz,     // (B,3,N)
    const float* __restrict__ newXyz,  // ws: (B,NP,3)
    int* __restrict__ ballIdx)         // ws: (B*NP, NS)
{
  const int lane = threadIdx.x & 31;
  const int wv   = threadIdx.x >> 5;
  const int gw   = blockIdx.x * 4 + wv;   // (b,s) group id
  const int b    = gw >> 9;
  const int s    = gw & 511;

  const float* xb = xyz + (size_t)b * 3 * NN;
  const float qx = newXyz[((size_t)b * NP + s) * 3 + 0];
  const float qy = newXyz[((size_t)b * NP + s) * 3 + 1];
  const float qz = newXyz[((size_t)b * NP + s) * 3 + 2];

  __shared__ int buf[4][NS];
  int* mybuf = buf[wv];

  int count = 0;
  for (int p0 = 0; p0 < NN && count < NS; p0 += 32) {
    const int p = p0 + lane;
    const float dx = xb[p] - qx;
    const float dy = xb[NN + p] - qy;
    const float dz = xb[2 * NN + p] - qz;
    const bool hit = (dx * dx + dy * dy + dz * dz) <= RAD2;
    const unsigned mask = (unsigned)__ballot(hit);
    const int pre = __popc(mask & ((1u << lane) - 1u));
    if (hit && (count + pre) < NS) mybuf[count + pre] = p;
    count += __popc(mask);
  }
  asm volatile("" ::: "memory");   // keep LDS store->load ordered (intra-wave)

  const int valid = (count < NS) ? count : NS;
  int v;
  if (valid == 0) {
    v = NN - 1;                     // reference index n clips to n-1 on gather
  } else {
    const int first = mybuf[0];
    v = (lane < valid) ? mybuf[lane] : first;
  }
  ballIdx[(size_t)gw * NS + lane] = v;
}

// ---------------------------------------------------------------------------
// A-fragment assembly from one LDS activation row (16B-aligned vector loads).
// Halves h=0..7 hold K = base+hi*8+(0..7); h=8..15 hold K = base+16+hi*8+(0..7)
// ---------------------------------------------------------------------------
__device__ __forceinline__ v16h ldA(const _Float16* rowp, int hi) {
  const v8h p0 = *(const v8h*)(rowp + hi * 8);
  const v8h p1 = *(const v8h*)(rowp + 16 + hi * 8);
  v16h a;
#pragma unroll
  for (int h = 0; h < 8; ++h) { a[h] = p0[h]; a[8 + h] = p1[h]; }
  return a;
}

// ---------------------------------------------------------------------------
// Kernel 3: fused gather + 3-layer pointwise MLP (WMMA f16->f32) + BN + ReLU
// + max over the 32 samples. One wave32 per (b,s) group (32 rows = 2 M-tiles).
// Fragment layouts per CDNA5 ISA 7.12.2:
//   A 16x32 f16: lane l -> row (l&15); half h -> K=(h>>3)*16+(l>>4)*8+(h&7)
//   B 32x16 f16: lane l -> col (l&15); half h -> K=(h>>3)*16+(l>>4)*8+(h&7)  (pre-swizzled)
//   C/D 16x16 f32: c[r] -> (M=r+8*(l>>4), N=(l&15))
// ---------------------------------------------------------------------------
__global__ __launch_bounds__(128) void mlp_kernel(
    const float*    __restrict__ xyz,  const float* __restrict__ pts,
    const _Float16* __restrict__ f0,  const _Float16* __restrict__ f1,
    const _Float16* __restrict__ f2,
    const float2*   __restrict__ bn0, const float2* __restrict__ bn1,
    const float2*   __restrict__ bn2,
    const int*      __restrict__ ballIdx,   // (B*NP, NS)
    const float*    __restrict__ newXyz,    // (B,NP,3)
    float*          __restrict__ out2)      // (B,128,NP)
{
  // per-wave staging; row stride 72 halves = 144B (16B-aligned)
  __shared__ __attribute__((aligned(16))) _Float16 act[4][32][72];

  const int lane = threadIdx.x & 31;
  const int wv   = threadIdx.x >> 5;
  const int gw   = blockIdx.x * 4 + wv;
  const int b    = gw >> 9;
  const int s    = gw & 511;
  const int lo   = lane & 15;
  const int hi   = lane >> 4;

  // ---------------- Layer 0: gather (9 ch) -> 64, K padded to 32 ------------
  v16h aIn[2];
#pragma unroll
  for (int t = 0; t < 2; ++t) {
    const int row  = t * 16 + lo;
    const int pidx = ballIdx[(size_t)gw * NS + row];
    float f[9];
#pragma unroll
    for (int c = 0; c < 3; ++c)
      f[c] = xyz[((size_t)b * 3 + c) * NN + pidx] -
             newXyz[((size_t)b * NP + s) * 3 + c];
#pragma unroll
    for (int c = 0; c < 6; ++c)
      f[3 + c] = pts[((size_t)b * 6 + c) * NN + pidx];
#pragma unroll
    for (int h = 0; h < 16; ++h) {
      const int k = ((h >> 3) << 4) + (hi << 3) + (h & 7);
      aIn[t][h] = (k < 9) ? (_Float16)f[k] : (_Float16)0.f;
    }
  }
#pragma unroll
  for (int j = 0; j < 4; ++j) {
    const int oc = j * 16 + lo;
    const v16h bf = *(const v16h*)(f0 + ((size_t)j * 32 + lane) * 16);
    const float2 ss = bn0[oc];
#pragma unroll
    for (int t = 0; t < 2; ++t) {
      v8f acc = {};
      acc = wmma_f16(aIn[t], bf, acc);
#pragma unroll
      for (int r = 0; r < 8; ++r) {
        const float y = fmaxf(acc[r] * ss.x + ss.y, 0.f);
        act[wv][t * 16 + r + hi * 8][oc] = (_Float16)y;
      }
    }
  }
  asm volatile("" ::: "memory");

  // ---------------- Layer 1: 64 -> 64 (2 K-tiles) ---------------------------
  v16h a0[2], a1[2];
#pragma unroll
  for (int t = 0; t < 2; ++t) {
    const _Float16* rowp = &act[wv][t * 16 + lo][0];
    a0[t] = ldA(rowp, hi);
    a1[t] = ldA(rowp + 32, hi);
  }
  asm volatile("" ::: "memory");   // all A reads before in-place stores
#pragma unroll
  for (int j = 0; j < 4; ++j) {
    const int oc = j * 16 + lo;
    const v16h bf0 = *(const v16h*)(f1 + ((size_t)(j * 2 + 0) * 32 + lane) * 16);
    const v16h bf1 = *(const v16h*)(f1 + ((size_t)(j * 2 + 1) * 32 + lane) * 16);
    const float2 ss = bn1[oc];
#pragma unroll
    for (int t = 0; t < 2; ++t) {
      v8f acc = {};
      acc = wmma_f16(a0[t], bf0, acc);
      acc = wmma_f16(a1[t], bf1, acc);
#pragma unroll
      for (int r = 0; r < 8; ++r) {
        const float y = fmaxf(acc[r] * ss.x + ss.y, 0.f);
        act[wv][t * 16 + r + hi * 8][oc] = (_Float16)y;
      }
    }
  }
  asm volatile("" ::: "memory");

  // ---------------- Layer 2: 64 -> 128, fused k-max -------------------------
#pragma unroll
  for (int t = 0; t < 2; ++t) {
    const _Float16* rowp = &act[wv][t * 16 + lo][0];
    a0[t] = ldA(rowp, hi);
    a1[t] = ldA(rowp + 32, hi);
  }
  float pm[8];
#pragma unroll
  for (int j = 0; j < 8; ++j) pm[j] = 0.f;   // post-ReLU values are >= 0

#pragma unroll
  for (int j = 0; j < 8; ++j) {
    const int oc = j * 16 + lo;
    const v16h bf0 = *(const v16h*)(f2 + ((size_t)(j * 2 + 0) * 32 + lane) * 16);
    const v16h bf1 = *(const v16h*)(f2 + ((size_t)(j * 2 + 1) * 32 + lane) * 16);
    const float2 ss = bn2[oc];
#pragma unroll
    for (int t = 0; t < 2; ++t) {
      v8f acc = {};
      acc = wmma_f16(a0[t], bf0, acc);
      acc = wmma_f16(a1[t], bf1, acc);
#pragma unroll
      for (int r = 0; r < 8; ++r) {
        const float y = fmaxf(acc[r] * ss.x + ss.y, 0.f);
        pm[j] = fmaxf(pm[j], y);
      }
    }
  }

  // combine the two row-halves (lanes l and l^16 hold same channel)
#pragma unroll
  for (int j = 0; j < 8; ++j) {
    const float o = fmaxf(pm[j], __shfl_xor(pm[j], 16, 32));
    if (lane < 16)
      out2[((size_t)b * 128 + (j * 16 + lane)) * NP + s] = o;
  }
}

// ---------------------------------------------------------------------------
extern "C" void kernel_launch(void* const* d_in, const int* in_sizes, int n_in,
                              void* d_out, int out_size, void* d_ws, size_t ws_size,
                              hipStream_t stream) {
  const float* xyz = (const float*)d_in[0];
  const float* pts = (const float*)d_in[1];
  const float* w0  = (const float*)d_in[2];
  const float* cb0 = (const float*)d_in[3];
  const float* g0  = (const float*)d_in[4];
  const float* be0 = (const float*)d_in[5];
  const float* m0  = (const float*)d_in[6];
  const float* v0  = (const float*)d_in[7];
  const float* w1  = (const float*)d_in[8];
  const float* cb1 = (const float*)d_in[9];
  const float* g1  = (const float*)d_in[10];
  const float* be1 = (const float*)d_in[11];
  const float* m1  = (const float*)d_in[12];
  const float* v1  = (const float*)d_in[13];
  const float* w2  = (const float*)d_in[14];
  const float* cb2 = (const float*)d_in[15];
  const float* g2  = (const float*)d_in[16];
  const float* be2 = (const float*)d_in[17];
  const float* m2  = (const float*)d_in[18];
  const float* v2  = (const float*)d_in[19];

  float* out1 = (float*)d_out;                    // (B,3,NP)
  float* out2 = out1 + (size_t)BB * 3 * NP;       // (B,128,NP)

  // workspace layout (all sections 32B-aligned)
  char* ws = (char*)d_ws;
  float*    newXyz  = (float*)ws;                         ws += (size_t)BB * NP * 3 * sizeof(float);     // 98304 B
  int*      ballIdx = (int*)ws;                           ws += (size_t)BB * NP * NS * sizeof(int);      // 1 MiB
  _Float16* f0      = (_Float16*)ws;                      ws += 4  * 32 * 16 * sizeof(_Float16);         // 4 KiB
  _Float16* f1      = (_Float16*)ws;                      ws += 8  * 32 * 16 * sizeof(_Float16);         // 8 KiB
  _Float16* f2      = (_Float16*)ws;                      ws += 16 * 32 * 16 * sizeof(_Float16);         // 16 KiB
  float2*   bn0     = (float2*)ws;                        ws += 64  * sizeof(float2);
  float2*   bn1     = (float2*)ws;                        ws += 64  * sizeof(float2);
  float2*   bn2     = (float2*)ws;                        ws += 128 * sizeof(float2);

  prep_kernel<<<1, 256, 0, stream>>>(
      w0, cb0, g0, be0, m0, v0,
      w1, cb1, g1, be1, m1, v1,
      w2, cb2, g2, be2, m2, v2,
      f0, f1, f2, bn0, bn1, bn2);

  fps_kernel<<<BB, 256, 0, stream>>>(xyz, newXyz, out1);

  const int ngroups = BB * NP;                    // 8192 (b,s) groups
  ballq_kernel<<<ngroups / 4, 128, 0, stream>>>(xyz, newXyz, ballIdx);

  mlp_kernel<<<ngroups / 4, 128, 0, stream>>>(
      xyz, pts, f0, f1, f2, bn0, bn1, bn2, ballIdx, newXyz, out2);
}
